// GPG_66726611910809
// MI455X (gfx1250) — matrix operand order repaired and checked by reading, not compiled, not run on hardware
//
#include <hip/hip_runtime.h>
#include <math.h>

// ---------------------------------------------------------------------------
// Problem constants
#define CFEAT   512
#define NPIX    65536          // 256*256
#define NBINS   50
#define NROWS   64             // 51 valid rows (glob + 50 local protos), padded
// ---------------------------------------------------------------------------

typedef __attribute__((ext_vector_type(16))) __bf16 v16bf;
typedef __attribute__((ext_vector_type(8)))  __bf16 v8bf;
typedef __attribute__((ext_vector_type(8)))  float  v8f;

// ---- WMMA helpers (CDNA5 wave32, 16x16x32 bf16 -> f32) --------------------
__device__ __forceinline__ v8f wmma_bf16(v16bf a, v16bf b, v8f c) {
  // 8 args: (neg_a, A, neg_b, B, c_mod, C, reuse_a, reuse_b)
  return __builtin_amdgcn_wmma_f32_16x16x32_bf16(false, a, false, b,
                                                 (short)0, c, false, false);
}

// A tile (16x32 bf16) from row-major bf16 LDS buffer.
// Lane L: row M = mbase + (L&15); K = kb + half*16 + e (contiguous 16)
__device__ __forceinline__ v16bf ldA(const __bf16* base, int stride,
                                     int mbase, int kbase, int lane) {
  int half = lane >> 4;
  const __bf16* rp = base + (mbase + (lane & 15)) * stride;
  v8bf lo = *(const v8bf*)(rp + kbase + half * 16);
  v8bf hi = *(const v8bf*)(rp + kbase + half * 16 + 8);
  v16bf a;
#pragma unroll
  for (int e = 0; e < 8; ++e) { a[e] = lo[e]; a[e + 8] = hi[e]; }
  return a;
}

// B tile (32x16 bf16) where storage is "n-major": buf[n][k] == B[k][n].
// Lane L: col N = nbase + (L&15); K = kbase + (L>>4)*16 + e (contiguous 16).
__device__ __forceinline__ v16bf ldBn(const __bf16* base, int stride,
                                      int nbase, int kbase, int lane) {
  const __bf16* rp = base + (nbase + (lane & 15)) * stride
                          + kbase + (lane >> 4) * 16;
  return *(const v16bf*)rp;
}

// Same B tile but sourced from a global row-major f32 weight matrix W[n][k].
__device__ __forceinline__ v16bf ldBg(const float* base, int stride,
                                      int nbase, int kbase, int lane) {
  const float* rp = base + (size_t)(nbase + (lane & 15)) * stride
                         + kbase + (lane >> 4) * 16;
  v16bf b;
#pragma unroll
  for (int e = 0; e < 16; ++e) b[e] = (__bf16)rp[e];
  return b;
}

// ---------------------------------------------------------------------------
// Kernel 1: per-block (1024 px) exclusive prefix of the mask
// ---------------------------------------------------------------------------
__global__ void k_scan_local(const int* __restrict__ msk,
                             int* __restrict__ localrank,
                             int* __restrict__ blockTotal) {
  __shared__ int tsum[256];
  int t = threadIdx.x;
  int base = blockIdx.x * 1024 + t * 4;
  int4 m = *(const int4*)(msk + base);
  int s = m.x + m.y + m.z + m.w;
  tsum[t] = s;
  __syncthreads();
  for (int off = 1; off < 256; off <<= 1) {
    int v = tsum[t];
    int u = (t >= off) ? tsum[t - off] : 0;
    __syncthreads();
    tsum[t] = v + u;
    __syncthreads();
  }
  int excl = tsum[t] - s;
  int4 r;
  r.x = excl; r.y = excl + m.x; r.z = r.y + m.y; r.w = r.z + m.z;
  *(int4*)(localrank + base) = r;
  if (t == 255) blockTotal[blockIdx.x] = tsum[255];
}

// ---------------------------------------------------------------------------
// Kernel 2: scan the 64 block totals, total fg count L, bin reciprocals
// scal: [0]=L (int bits), [1]=Lf, [2]=1/(Lf+1e-8)
// ---------------------------------------------------------------------------
__global__ void k_scan_blocks(const int* __restrict__ blockTotal,
                              int* __restrict__ blockOffset,
                              float* __restrict__ scal,
                              float* __restrict__ recip) {
  __shared__ int s[64];
  __shared__ int Ls;
  int t = threadIdx.x;             // 64 threads
  s[t] = blockTotal[t];
  __syncthreads();
  if (t == 0) {
    int acc = 0;
    for (int i = 0; i < 64; ++i) { int v = s[i]; s[i] = acc; acc += v; }
    Ls = acc;
    ((int*)scal)[0] = acc;
    float Lf = (float)acc;
    scal[1] = Lf;
    scal[2] = 1.0f / (Lf + 1e-8f);
  }
  __syncthreads();
  blockOffset[t] = s[t];
  int L = Ls;
  if (t < NBINS) {
    long long Ll = (long long)L;
    int st = (int)(((long long)t * Ll) / NBINS);
    int en = (int)((((long long)(t + 1) * Ll) + NBINS - 1) / NBINS);
    int cnt = en - st; if (cnt < 1) cnt = 1;
    recip[t] = 1.0f / (float)cnt;
  }
}

// ---------------------------------------------------------------------------
// Kernel 3: single pass over fts (128 MB) -> global sum + adaptive-pool bins
// grid (64 pixel tiles, 4 channel groups), 256 threads
// ---------------------------------------------------------------------------
__global__ void k_accum(const float* __restrict__ fts,
                        const int* __restrict__ msk,
                        const int* __restrict__ localrank,
                        const int* __restrict__ blockOffset,
                        const float* __restrict__ scal,
                        float* __restrict__ glob_sum,
                        float* __restrict__ bin_sum) {
  __shared__ float bins[NBINS][128];
  __shared__ float glob[128];
  int t = threadIdx.x;
  int tile = blockIdx.x;           // pixel tile (1024 px)
  int cbase = blockIdx.y * 128;    // channel group
  for (int i = t; i < NBINS * 128; i += 256) ((float*)bins)[i] = 0.f;
  if (t < 128) glob[t] = 0.f;
  __syncthreads();

  int L = ((const int*)scal)[0];
  if (L < 1) L = 1;
  int pbase = tile * 1024;
  int boff = blockOffset[tile];

  int  pm[4], b0[4];
  bool h2[4];
#pragma unroll
  for (int j = 0; j < 4; ++j) {
    int p = pbase + t + 256 * j;
    pm[j] = msk[p];
    int r = boff + localrank[p];
    int i0 = (NBINS * r) / L;
    if (i0 > NBINS - 1) i0 = NBINS - 1;
    b0[j] = i0;
    h2[j] = (i0 + 1 < NBINS) && ((i0 + 1) * L < NBINS * (r + 1));
  }

  for (int c = 0; c < 128; ++c) {
    const float* fp = fts + (size_t)(cbase + c) * NPIX + pbase;
    float gacc = 0.f;
#pragma unroll
    for (int j = 0; j < 4; ++j) {
      float v = fp[t + 256 * j];
      if (pm[j]) {
        gacc += v;
        atomicAdd(&bins[b0[j]][c], v);
        if (h2[j]) atomicAdd(&bins[b0[j] + 1][c], v);
      }
    }
    for (int off = 16; off > 0; off >>= 1) gacc += __shfl_down(gacc, off);
    if ((t & 31) == 0) atomicAdd(&glob[c], gacc);
  }
  __syncthreads();
  for (int i = t; i < NBINS * 128; i += 256) {
    int bi = i / 128, ci = i % 128;
    atomicAdd(&bin_sum[bi * CFEAT + cbase + ci], bins[bi][ci]);
  }
  if (t < 128) atomicAdd(&glob_sum[cbase + t], glob[t]);
}

// ---------------------------------------------------------------------------
// Kernel 4: assemble H [64 x 512] (row 0 = global proto, rows 1..50 = bins)
// ---------------------------------------------------------------------------
__global__ void k_build_H(const float* __restrict__ glob_sum,
                          const float* __restrict__ bin_sum,
                          const float* __restrict__ scal,
                          const float* __restrict__ recip,
                          float* __restrict__ H) {
  int row = blockIdx.x, c = threadIdx.x;   // 64 x 512
  float v;
  if (row == 0)        v = glob_sum[c] * scal[2];
  else if (row <= NBINS) v = bin_sum[(row - 1) * CFEAT + c] * recip[row - 1];
  else                 v = 0.f;
  H[row * CFEAT + c] = v;
}

// ---------------------------------------------------------------------------
// Kernel 5: one GCN layer, WMMA bf16.  1 block, 1024 threads (32 waves).
//   S   = Hn @ Hn^T            (16 tiles, K=512)
//   adj = softmax(relu(S)) * diag(norm)   (folds un-normalization)
//   agg = adj @ Hn             (128 tiles, K=64)
//   H  += relu(agg @ W^T + b)  (128 tiles, K=512, B streamed from global W)
// ---------------------------------------------------------------------------
__global__ __launch_bounds__(1024) void k_gcn(float* __restrict__ H,
                                              const float* __restrict__ W,
                                              const float* __restrict__ bias) {
  __shared__ __bf16 buf1[NROWS * CFEAT];   // Hn row-major, later agg (bf16)
  __shared__ __bf16 buf2[CFEAT * NROWS];   // Hn channel-major (HnT)
  __shared__ float  adjf[NROWS * NROWS];
  __shared__ __bf16 adjb[NROWS * NROWS];
  __shared__ float  norms[NROWS];
  __shared__ float  part[NROWS * 16];

  int t = threadIdx.x, lane = t & 31, w = t >> 5;

  // --- row norms ---
  {
    int r = t >> 4, seg = t & 15;
    const float* hp = H + r * CFEAT + seg * 32;
    float s = 0.f;
#pragma unroll
    for (int i = 0; i < 32; ++i) { float x = hp[i]; s += x * x; }
    part[r * 16 + seg] = s;
  }
  __syncthreads();
  if (t < NROWS) {
    float s = 0.f;
    for (int i = 0; i < 16; ++i) s += part[t * 16 + i];
    float n = sqrtf(s);
    norms[t] = (n > 1e-12f) ? n : 1e-12f;
  }
  __syncthreads();

  // --- buf1 = Hn (row-major bf16), buf2 = Hn^T (channel-major bf16) ---
  for (int i = t; i < NROWS * CFEAT; i += 1024) {
    int r = i >> 9;
    buf1[i] = (__bf16)(H[i] / norms[r]);
  }
  for (int i = t; i < CFEAT * NROWS; i += 1024) {
    int ch = i >> 6, r = i & 63;
    buf2[i] = (__bf16)(H[r * CFEAT + ch] / norms[r]);
  }
  __syncthreads();

  // --- Gram S = Hn @ Hn^T : 4x4 tiles, waves 0..15 ---
  if (w < 16) {
    int mt = w >> 2, nt = w & 3;
    v8f acc = {};
    for (int kk = 0; kk < CFEAT; kk += 32) {
      v16bf a = ldA(buf1, CFEAT, mt * 16, kk, lane);
      v16bf b = ldBn(buf1, CFEAT, nt * 16, kk, lane);   // B[k][n] = Hn[n][k]
      acc = wmma_bf16(a, b, acc);
    }
    int half = lane >> 4, n = nt * 16 + (lane & 15);
#pragma unroll
    for (int r8 = 0; r8 < 8; ++r8)
      adjf[(mt * 16 + r8 + half * 8) * NROWS + n] = acc[r8];
  }
  __syncthreads();

  // --- adj = softmax(relu(S)) over 51 cols, fold * norm[k] -> adjb ---
  if (t < NROWS) {
    if (t < 51) {
      float mx = 0.f;
      for (int j = 0; j < 51; ++j) {
        float x = adjf[t * NROWS + j];
        x = x > 0.f ? x : 0.f;
        adjf[t * NROWS + j] = x;
        if (x > mx) mx = x;
      }
      float s = 0.f;
      for (int j = 0; j < 51; ++j) {
        float e = __expf(adjf[t * NROWS + j] - mx);
        adjf[t * NROWS + j] = e; s += e;
      }
      float inv = 1.f / s;
      for (int j = 0; j < 51; ++j)
        adjb[t * NROWS + j] = (__bf16)(adjf[t * NROWS + j] * inv * norms[j]);
      for (int j = 51; j < NROWS; ++j) adjb[t * NROWS + j] = (__bf16)0.f;
    } else {
      for (int j = 0; j < NROWS; ++j) adjb[t * NROWS + j] = (__bf16)0.f;
    }
  }
  __syncthreads();

  // --- agg = (adj*diag(norm)) @ Hn : 4x32 tiles, 4 per wave; out -> buf1 ---
  for (int tt = 0; tt < 4; ++tt) {
    int tile = w * 4 + tt, mt = tile >> 5, nt = tile & 31;
    v8f acc = {};
    for (int kk = 0; kk < NROWS; kk += 32) {
      v16bf a = ldA(adjb, NROWS, mt * 16, kk, lane);
      v16bf b = ldBn(buf2, NROWS, nt * 16, kk, lane);   // B[k][n] = Hn[k][n]
      acc = wmma_bf16(a, b, acc);
    }
    int half = lane >> 4, n = nt * 16 + (lane & 15);
#pragma unroll
    for (int r8 = 0; r8 < 8; ++r8)
      buf1[(mt * 16 + r8 + half * 8) * CFEAT + n] = (__bf16)acc[r8];
  }
  __syncthreads();

  // --- H += relu(agg @ W^T + b) : B streamed from global W[j][k] ---
  for (int tt = 0; tt < 4; ++tt) {
    int tile = w * 4 + tt, mt = tile >> 5, nt = tile & 31;
    v8f acc = {};
    for (int kk = 0; kk < CFEAT; kk += 32) {
      v16bf a = ldA(buf1, CFEAT, mt * 16, kk, lane);
      v16bf b = ldBg(W, CFEAT, nt * 16, kk, lane);
      acc = wmma_bf16(a, b, acc);
    }
    int half = lane >> 4, n = nt * 16 + (lane & 15);
    float bn = bias[n];
#pragma unroll
    for (int r8 = 0; r8 < 8; ++r8) {
      int m = mt * 16 + r8 + half * 8;
      if (m < 51) {
        float z = acc[r8] + bn;
        z = z > 0.f ? z : 0.f;
        H[m * CFEAT + n] += z;
      }
    }
  }
}

// ---------------------------------------------------------------------------
// Kernel 6: attention head. k/v projections via WMMA; the rest is tiny VALU.
// ---------------------------------------------------------------------------
__global__ __launch_bounds__(1024) void k_attn(
    const float* __restrict__ H,
    const float* __restrict__ qw, const float* __restrict__ qb,
    const float* __restrict__ kw, const float* __restrict__ kb,
    const float* __restrict__ vw, const float* __restrict__ vb,
    const float* __restrict__ mw, const float* __restrict__ mb,
    const float* __restrict__ gamma, const float* __restrict__ alpha,
    const float* __restrict__ glob_sum, const float* __restrict__ scal,
    float* __restrict__ out) {
  __shared__ __bf16 nodes[NROWS * CFEAT];   // H[1..50] padded, bf16 row-major
  __shared__ float  kf[NROWS * 256];
  __shared__ float  qv[256];
  __shared__ float  qn[CFEAT];
  __shared__ float  attnw[NROWS];
  __shared__ float  attnv[CFEAT];

  int t = threadIdx.x, lane = t & 31, w = t >> 5;

  for (int i = t; i < NROWS * CFEAT; i += 1024) {
    int r = i >> 9, c = i & 511;
    nodes[i] = (__bf16)((r < NBINS) ? H[(r + 1) * CFEAT + c] : 0.f);
  }
  if (t < CFEAT) { qn[t] = H[t]; attnv[t] = 0.f; }
  __syncthreads();

  // q = H0 @ qw^T + qb
  if (t < 256) {
    const float* wr = qw + t * CFEAT;
    float s = qb[t];
    for (int c = 0; c < CFEAT; ++c) s += qn[c] * wr[c];
    qv[t] = s;
  }

  // k = nodes @ kw^T + kb : 4x16 tiles, 2 per wave
  for (int tt = 0; tt < 2; ++tt) {
    int tile = w * 2 + tt, mt = tile >> 4, nt = tile & 15;
    v8f acc = {};
    for (int kk = 0; kk < CFEAT; kk += 32) {
      v16bf a = ldA(nodes, CFEAT, mt * 16, kk, lane);
      v16bf b = ldBg(kw, CFEAT, nt * 16, kk, lane);
      acc = wmma_bf16(a, b, acc);
    }
    int half = lane >> 4, n = nt * 16 + (lane & 15);
    float bn = kb[n];
#pragma unroll
    for (int r8 = 0; r8 < 8; ++r8)
      kf[(mt * 16 + r8 + half * 8) * 256 + n] = acc[r8] + bn;
  }
  __syncthreads();

  // attn logits + softmax over 50 nodes
  if (t < NBINS) {
    const float* kr = kf + t * 256;
    float s = 0.f;
    for (int j = 0; j < 256; ++j) s += qv[j] * kr[j];
    attnw[t] = s * 0.0625f;                 // 1/sqrt(256)
  }
  __syncthreads();
  if (t == 0) {
    float mx = attnw[0];
    for (int i = 1; i < NBINS; ++i) if (attnw[i] > mx) mx = attnw[i];
    float s = 0.f;
    for (int i = 0; i < NBINS; ++i) { float e = __expf(attnw[i] - mx); attnw[i] = e; s += e; }
    float inv = 1.f / s;
    for (int i = 0; i < NBINS; ++i) attnw[i] *= inv;
    for (int i = NBINS; i < NROWS; ++i) attnw[i] = 0.f;
  }
  __syncthreads();

  // v = nodes @ vw^T + vb, fused with attn @ v  (accumulate into attnv)
  for (int tt = 0; tt < 4; ++tt) {
    int tile = w * 4 + tt, mt = tile >> 5, nt = tile & 31;
    v8f acc = {};
    for (int kk = 0; kk < CFEAT; kk += 32) {
      v16bf a = ldA(nodes, CFEAT, mt * 16, kk, lane);
      v16bf b = ldBg(vw, CFEAT, nt * 16, kk, lane);
      acc = wmma_bf16(a, b, acc);
    }
    int half = lane >> 4, n = nt * 16 + (lane & 15);
    float bn = vb[n];
#pragma unroll
    for (int r8 = 0; r8 < 8; ++r8) {
      int m = mt * 16 + r8 + half * 8;
      if (m < NBINS) atomicAdd(&attnv[n], attnw[m] * (acc[r8] + bn));
    }
  }
  __syncthreads();

  // map + final blend
  if (t < CFEAT) {
    const float* wr = mw + t * CFEAT;
    float s = 0.f;
    for (int c = 0; c < CFEAT; ++c) s += attnv[c] * wr[c];
    float fin  = qn[t] + gamma[0] * (s + mb[t]);
    float wsig = 1.f / (1.f + __expf(-alpha[0]));
    float gl   = glob_sum[t] * scal[2];
    out[t] = gl * (1.f - wsig) + fin * wsig;
  }
}

// ---------------------------------------------------------------------------
// Host launcher
// ---------------------------------------------------------------------------
extern "C" void kernel_launch(void* const* d_in, const int* in_sizes, int n_in,
                              void* d_out, int out_size, void* d_ws, size_t ws_size,
                              hipStream_t stream) {
  const float* fts   = (const float*)d_in[0];
  const int*   msk   = (const int*)  d_in[1];
  const float* gcn_w = (const float*)d_in[2];
  const float* gcn_b = (const float*)d_in[3];
  const float* q_w   = (const float*)d_in[4];
  const float* q_b   = (const float*)d_in[5];
  const float* k_w   = (const float*)d_in[6];
  const float* k_b   = (const float*)d_in[7];
  const float* v_w   = (const float*)d_in[8];
  const float* v_b   = (const float*)d_in[9];
  const float* map_w = (const float*)d_in[10];
  const float* map_b = (const float*)d_in[11];
  const float* gamma = (const float*)d_in[12];
  const float* alpha = (const float*)d_in[13];

  char* ws = (char*)d_ws;
  int*   localrank   = (int*)  (ws + 0);        // 262144 B
  int*   blockTotal  = (int*)  (ws + 262144);   // 256 B
  int*   blockOffset = (int*)  (ws + 262400);   // 256 B
  float* scal        = (float*)(ws + 262656);   // 256 B
  float* recip       = (float*)(ws + 262912);   // 256 B
  float* glob_sum    = (float*)(ws + 263168);   // 2048 B
  float* bin_sum     = (float*)(ws + 265216);   // 102400 B
  float* H           = (float*)(ws + 367616);   // 131072 B

  hipMemsetAsync(glob_sum, 0, 2048 + NBINS * CFEAT * sizeof(float), stream);

  k_scan_local <<<64, 256, 0, stream>>>(msk, localrank, blockTotal);
  k_scan_blocks<<<1, 64, 0, stream>>>(blockTotal, blockOffset, scal, recip);
  dim3 g3(64, 4);
  k_accum      <<<g3, 256, 0, stream>>>(fts, msk, localrank, blockOffset, scal,
                                        glob_sum, bin_sum);
  k_build_H    <<<64, 512, 0, stream>>>(glob_sum, bin_sum, scal, recip, H);
  for (int l = 0; l < 2; ++l)
    k_gcn      <<<1, 1024, 0, stream>>>(H, gcn_w + (size_t)l * CFEAT * CFEAT,
                                        gcn_b + (size_t)l * CFEAT);
  k_attn       <<<1, 1024, 0, stream>>>(H, q_w, q_b, k_w, k_b, v_w, v_b,
                                        map_w, map_b, gamma, alpha,
                                        glob_sum, scal, (float*)d_out);
}
